// NRGTrans_36326833389930
// MI455X (gfx1250) — matrix operand order repaired and checked by reading, not compiled, not run on hardware
//
#include <hip/hip_runtime.h>
#include <math.h>

// ---------------- model dims ----------------
#define NB_B   64
#define WIN    100
#define ENC_IN 38
#define C_OUT  38
#define DMODEL 512
#define NHEAD  8
#define DHEAD  64
#define MTOK   (NB_B * WIN)                 // 6400 token rows
#define ATTN1  (NB_B * NHEAD * WIN * WIN)   // 5,120,000 per (B,H,100,100) tensor

typedef __attribute__((ext_vector_type(16))) __bf16        v16bf;
typedef __attribute__((ext_vector_type(4)))  __bf16        v4bf;
typedef __attribute__((ext_vector_type(8)))  float         v8f;
typedef __attribute__((ext_vector_type(4)))  unsigned int  uint4v;

static inline unsigned cdiv(size_t a, unsigned b) { return (unsigned)((a + b - 1) / b); }

// =====================================================================
// WMMA GEMM v3:  C[M,N] = epilogue(A[M,K] @ B[K,N] + bias[N])
// A,B,C row-major f32.  bf16 WMMA, f32 accumulate.
// LDS tiles in bf16 *fragment-major* order -> inner loop is pure
// ds_load_b128 + v_wmma.  Staging batches global loads into registers
// before convert+ds_store so HBM latency is amortized, not serialized.
// Block = 256 threads (8 wave32); block tile 256x64; BK=64 (2 chunks);
// wave tile 32x64 (2 A frags x 4 B frags -> 16 WMMA per stage).
// Requirements: M % 256 == 0, K % 64 == 0; N arbitrary (guarded path).
// EPI: 0=none, 1=relu, 2=exact gelu
// =====================================================================
#define GBM 256
#define GBN 64
#define GPAD 40   // bf16 row stride: 80B = 5*16B -> aligned b128 frags, bank spread

__device__ __forceinline__ v16bf load_frag16(const __bf16* p) {
    struct BF16x16 { uint4v lo, hi; } t;
    t.lo = *(const uint4v*)p;
    t.hi = *(const uint4v*)(p + 8);
    return __builtin_bit_cast(v16bf, t);
}

template <int EPI>
__global__ __launch_bounds__(256)
void gemm_bias_wmma(const float* __restrict__ A, const float* __restrict__ Bm,
                    const float* __restrict__ bias, float* __restrict__ C,
                    int M, int N, int K) {
    // fragment-major bf16 tiles: [k-chunk][row|n][half*16 + slot] (+pad)
    __shared__ __bf16 Af[2][GBM][GPAD];
    __shared__ __bf16 Bf[2][GBN][GPAD];

    const int tid  = threadIdx.x;
    const int lane = tid & 31;
    const int wave = tid >> 5;                 // 0..7 -> 32-row stripe
    const int row0 = blockIdx.x * GBM;
    const int col0 = blockIdx.y * GBN;
    const bool bfast = (col0 + GBN <= N) && ((N & 3) == 0);   // wave-uniform

    v8f acc[2][4];
#pragma unroll
    for (int s = 0; s < 2; ++s)
#pragma unroll
        for (int nt = 0; nt < 4; ++nt)
#pragma unroll
            for (int r = 0; r < 8; ++r) acc[s][nt][r] = 0.0f;

    const int am   = lane & 15;                // A-frag row within 16-row tile
    const int half = lane >> 4;                // lane-half -> K sub-block
    const int bn   = lane & 15;                // B-frag column
    const int bk16 = (lane >> 4) * 16;         // B-frag K half

    for (int k0 = 0; k0 < K; k0 += 64) {
        __syncthreads();
        // ---- stage A tile 256x64 f32 -> bf16 fragment-major ----
        // batch 8 global float4 loads, then convert+store (amortized wait)
#pragma unroll
        for (int hblk = 0; hblk < 2; ++hblk) {
            float4 av[8];
#pragma unroll
            for (int it = 0; it < 8; ++it) {
                int idx = tid + 256 * (hblk * 8 + it);   // 0..4095 float4 slots
                int r   = idx >> 4;                      // 0..255
                int c4  = (idx & 15) * 4;                // 0..60
                av[it] = *(const float4*)(A + (size_t)(row0 + r) * K + k0 + c4);
            }
#pragma unroll
            for (int it = 0; it < 8; ++it) {
                int idx   = tid + 256 * (hblk * 8 + it);
                int r     = idx >> 4;
                int c4    = (idx & 15) * 4;
                int chunk = c4 >> 5;
                int kk    = c4 & 31;
                int h     = kk >> 4;
                int slot  = kk & 15;
                v4bf pk = { (__bf16)av[it].x, (__bf16)av[it].y,
                            (__bf16)av[it].z, (__bf16)av[it].w };
                *(v4bf*)&Af[chunk][r][h * 16 + slot] = pk;   // ds_store_b64
            }
        }
        // ---- stage B tile 64x64 f32 -> bf16 fragment-major ----
        if (bfast) {
            // vectorized: 4 x global_load_b128 per thread
            float4 bv[4];
#pragma unroll
            for (int it = 0; it < 4; ++it) {
                int idx = tid + 256 * it;          // 0..1023 float4 slots
                int r   = idx >> 4;                // K row 0..63
                int c4  = (idx & 15) * 4;
                bv[it] = *(const float4*)(Bm + (size_t)(k0 + r) * N + col0 + c4);
            }
#pragma unroll
            for (int it = 0; it < 4; ++it) {
                int idx   = tid + 256 * it;
                int r     = idx >> 4;
                int c4    = (idx & 15) * 4;
                int chunk = r >> 5;
                int kk    = r & 31;
                int sl    = (kk >> 4) * 16 + (kk & 15);
                Bf[chunk][c4 + 0][sl] = (__bf16)bv[it].x;
                Bf[chunk][c4 + 1][sl] = (__bf16)bv[it].y;
                Bf[chunk][c4 + 2][sl] = (__bf16)bv[it].z;
                Bf[chunk][c4 + 3][sl] = (__bf16)bv[it].w;
            }
        } else {
            // N-guarded scalar path (final projection, N=38)
            float bs[16];
#pragma unroll
            for (int it = 0; it < 16; ++it) {
                int idx = tid + 256 * it;          // 0..4095
                int r   = idx >> 6;
                int c   = idx & 63;
                int gc  = col0 + c;
                bs[it] = (gc < N) ? Bm[(size_t)(k0 + r) * N + gc] : 0.0f;
            }
#pragma unroll
            for (int it = 0; it < 16; ++it) {
                int idx   = tid + 256 * it;
                int r     = idx >> 6;
                int c     = idx & 63;
                int chunk = r >> 5;
                int kk    = r & 31;
                Bf[chunk][c][(kk >> 4) * 16 + (kk & 15)] = (__bf16)bs[it];
            }
        }
        // prefetch next A stripe into cache (global_prefetch_b8)
        if (k0 + 64 < K)
            __builtin_prefetch(A + (size_t)(row0 + tid) * K + k0 + 64, 0, 0);
        __syncthreads();

        // ---- compute: 2 chunks x (2 A-frags x 4 B-frags) ----
#pragma unroll
        for (int ch = 0; ch < 2; ++ch) {
            v16bf a0 = load_frag16(&Af[ch][wave * 32 + am][half * 16]);
            v16bf a1 = load_frag16(&Af[ch][wave * 32 + 16 + am][half * 16]);
#pragma unroll
            for (int nt = 0; nt < 4; ++nt) {
                v16bf b = load_frag16(&Bf[ch][nt * 16 + bn][bk16]);
                acc[0][nt] = __builtin_amdgcn_wmma_f32_16x16x32_bf16(
                    false, a0, false, b, (short)0, acc[0][nt], false, false);
                acc[1][nt] = __builtin_amdgcn_wmma_f32_16x16x32_bf16(
                    false, a1, false, b, (short)0, acc[1][nt], false, false);
            }
        }
    }

    // ---- epilogue + store (C/D layout: M = r + 8*(lane>>4), N = lane&15) ----
    const int chalf = lane >> 4;
    const int cn    = lane & 15;
#pragma unroll
    for (int s = 0; s < 2; ++s) {
#pragma unroll
        for (int nt = 0; nt < 4; ++nt) {
            int gcol = col0 + nt * 16 + cn;
            if (gcol >= N) continue;
            float bv = bias[gcol];
#pragma unroll
            for (int r = 0; r < 8; ++r) {
                int grow = row0 + wave * 32 + s * 16 + r + 8 * chalf;
                float v  = acc[s][nt][r] + bv;
                if (EPI == 1) v = fmaxf(v, 0.0f);
                if (EPI == 2) v = 0.5f * v * (1.0f + erff(v * 0.70710678118654752f));
                C[(size_t)grow * N + gcol] = v;
            }
        }
    }
}

// =====================================================================
// DataEmbedding: circular conv1d(k=3) token embedding + sinusoidal PE
// =====================================================================
__global__ __launch_bounds__(256)
void embed_kernel(const float* __restrict__ x, const float* __restrict__ ew,
                  float* __restrict__ enc) {
    int idx = blockIdx.x * blockDim.x + threadIdx.x;      // (b,t,d)
    if (idx >= MTOK * DMODEL) return;
    int d  = idx & (DMODEL - 1);
    int mt = idx >> 9;
    int t  = mt % WIN;
    int b  = mt / WIN;
    float s = 0.0f;
#pragma unroll
    for (int k = 0; k < 3; ++k) {
        int tt = t - 1 + k; tt = (tt + WIN) % WIN;        // wrap pad
        const float* xr = x + ((size_t)b * WIN + tt) * ENC_IN;
        const float* w  = ew + ((size_t)k * ENC_IN) * DMODEL + d;
        for (int c = 0; c < ENC_IN; ++c) s += xr[c] * w[(size_t)c * DMODEL];
    }
    int   i2  = d >> 1;
    float div = expf((float)(2 * i2) * (-9.210340371976184f / (float)DMODEL));
    float ang = (float)t * div;
    s += (d & 1) ? cosf(ang) : sinf(ang);
    enc[idx] = s;
}

// =====================================================================
// GaussianLayer: sigma(b,h,t) then prior/sigma broadcast to (B,H,L,L)
// =====================================================================
__global__ __launch_bounds__(256)
void gauss_sig_kernel(const float* __restrict__ enc, const float* __restrict__ wsig,
                      const float* __restrict__ bsig, float* __restrict__ sig) {
    int idx = blockIdx.x * blockDim.x + threadIdx.x;      // (b,h,t)
    if (idx >= NB_B * NHEAD * WIN) return;
    int t   = idx % WIN;
    int tmp = idx / WIN;
    int h   = tmp % NHEAD;
    int b   = tmp / NHEAD;
    const float* e = enc + ((size_t)b * WIN + t) * DMODEL;
    float s = bsig[h];
    for (int c = 0; c < DMODEL; ++c) s += e[c] * wsig[c * NHEAD + h];
    s = 1.0f / (1.0f + expf(-5.0f * s)) + 1e-5f;          // sigmoid(5x)+eps
    s = expf(s * 1.0986122886681098f) - 1.0f;             // 3^s - 1
    sig[idx] = s;
}

__global__ __launch_bounds__(256)
void prior_sigma_kernel(const float* __restrict__ sig, float* __restrict__ prior,
                        float* __restrict__ sigma_out) {
    size_t idx = (size_t)blockIdx.x * blockDim.x + threadIdx.x;
    if (idx >= (size_t)ATTN1) return;
    int j = (int)(idx % WIN); size_t t = idx / WIN;
    int i = (int)(t % WIN);   t /= WIN;
    int h = (int)(t % NHEAD);
    int b = (int)(t / NHEAD);
    float s  = sig[((size_t)b * NHEAD + h) * WIN + i];
    float dd = (float)abs(i - j);
    float pr = expf(-dd * dd / (2.0f * s * s)) * 0.3989422804014327f / s;
    prior[idx]     = pr;
    sigma_out[idx] = s;
}

// =====================================================================
// Attention pieces (patch size p; batches nb = 6400/p)
// =====================================================================
__global__ __launch_bounds__(256)
void scores_kernel(const float* __restrict__ Q, const float* __restrict__ Km,
                   const float* __restrict__ pre, float* __restrict__ out,
                   int p, int nb, float scale) {
    size_t idx   = (size_t)blockIdx.x * blockDim.x + threadIdx.x;
    size_t total = (size_t)nb * NHEAD * p * p;
    if (idx >= total) return;
    int j = (int)(idx % p); size_t t = idx / p;
    int i = (int)(t % p);   t /= p;
    int h  = (int)(t % NHEAD);
    int bp = (int)(t / NHEAD);
    const float* q = Q  + ((size_t)(bp * p + i) * DMODEL) + h * DHEAD;
    const float* k = Km + ((size_t)(bp * p + j) * DMODEL) + h * DHEAD;
    float s = 0.0f;
#pragma unroll 8
    for (int d = 0; d < DHEAD; ++d) s += q[d] * k[d];
    s *= scale;
    if (pre) s += pre[idx];
    out[idx] = s;
}

__global__ __launch_bounds__(256)
void softmax_kernel(float* __restrict__ buf, int rows, int p) {
    int row  = blockIdx.x * (blockDim.x >> 5) + (threadIdx.x >> 5);
    int lane = threadIdx.x & 31;
    if (row >= rows) return;
    float* r = buf + (size_t)row * p;
    float mx = -1e30f;
    for (int j = lane; j < p; j += 32) mx = fmaxf(mx, r[j]);
#pragma unroll
    for (int o = 16; o; o >>= 1) mx = fmaxf(mx, __shfl_xor(mx, o, 32));
    float sum = 0.0f;
    for (int j = lane; j < p; j += 32) { float e = expf(r[j] - mx); r[j] = e; sum += e; }
#pragma unroll
    for (int o = 16; o; o >>= 1) sum += __shfl_xor(sum, o, 32);
    float inv = 1.0f / sum;
    for (int j = lane; j < p; j += 32) r[j] *= inv;
}

__global__ __launch_bounds__(256)
void attn_apply_kernel(const float* __restrict__ attn, const float* __restrict__ V,
                       float* __restrict__ out, int p) {
    int idx = blockIdx.x * blockDim.x + threadIdx.x;      // (m, c)
    if (idx >= MTOK * DMODEL) return;
    int c    = idx & (DMODEL - 1);
    int mrow = idx >> 9;
    int h    = c >> 6;
    int bp   = mrow / p;
    int i    = mrow % p;
    const float* a = attn + (((size_t)bp * NHEAD + h) * p + i) * p;
    const float* v = V + (size_t)(bp * p) * DMODEL + c;
    float s = 0.0f;
    for (int j = 0; j < p; ++j) s += a[j] * v[(size_t)j * DMODEL];
    out[idx] = s;
}

// =====================================================================
// LayerNorm family (row length 512, one 256-thread / 8-wave block per row)
// =====================================================================
__device__ __forceinline__ float block_reduce_sum(float v, float* sm) {
    int lane = threadIdx.x & 31, w = threadIdx.x >> 5;
#pragma unroll
    for (int o = 16; o; o >>= 1) v += __shfl_xor(v, o, 32);
    if (lane == 0) sm[w] = v;
    __syncthreads();
    float r = (threadIdx.x < 8) ? sm[threadIdx.x] : 0.0f;
    if (w == 0) {
#pragma unroll
        for (int o = 4; o; o >>= 1) r += __shfl_xor(r, o, 32);
        if (lane == 0) sm[0] = r;
    }
    __syncthreads();
    r = sm[0];
    __syncthreads();
    return r;
}

__global__ __launch_bounds__(256)
void add_ln_kernel(const float* __restrict__ a, const float* __restrict__ b,
                   const float* __restrict__ g, const float* __restrict__ be,
                   float* __restrict__ out, int outStride) {
    __shared__ float sm[8];
    int row = blockIdx.x, t = threadIdx.x;
    const float* ar = a + (size_t)row * DMODEL;
    const float* br = b + (size_t)row * DMODEL;
    float x0 = ar[t] + br[t];
    float x1 = ar[t + 256] + br[t + 256];
    float mean = block_reduce_sum(x0 + x1, sm) * (1.0f / DMODEL);
    float d0 = x0 - mean, d1 = x1 - mean;
    float var = block_reduce_sum(d0 * d0 + d1 * d1, sm) * (1.0f / DMODEL);
    float rs = rsqrtf(var + 1e-5f);
    float* o = out + (size_t)row * outStride;
    o[t]       = d0 * rs * g[t]       + be[t];
    o[t + 256] = d1 * rs * g[t + 256] + be[t + 256];
}

__global__ __launch_bounds__(256)
void add_ln2_kernel(const float* __restrict__ a, const float* __restrict__ b,
                    const float* __restrict__ g1, const float* __restrict__ be1,
                    const float* __restrict__ g2, const float* __restrict__ be2,
                    float* __restrict__ out, int outStride) {
    __shared__ float sm[8];
    int row = blockIdx.x, t = threadIdx.x;
    const float* ar = a + (size_t)row * DMODEL;
    const float* br = b + (size_t)row * DMODEL;
    float x0 = ar[t] + br[t];
    float x1 = ar[t + 256] + br[t + 256];
    float mean = block_reduce_sum(x0 + x1, sm) * (1.0f / DMODEL);
    float d0 = x0 - mean, d1 = x1 - mean;
    float var = block_reduce_sum(d0 * d0 + d1 * d1, sm) * (1.0f / DMODEL);
    float rs = rsqrtf(var + 1e-5f);
    float y0 = d0 * rs * g1[t]       + be1[t];
    float y1 = d1 * rs * g1[t + 256] + be1[t + 256];
    float mean2 = block_reduce_sum(y0 + y1, sm) * (1.0f / DMODEL);
    float e0 = y0 - mean2, e1 = y1 - mean2;
    float var2 = block_reduce_sum(e0 * e0 + e1 * e1, sm) * (1.0f / DMODEL);
    float rs2 = rsqrtf(var2 + 1e-5f);
    float* o = out + (size_t)row * outStride;
    o[t]       = e0 * rs2 * g2[t]       + be2[t];
    o[t + 256] = e1 * rs2 * g2[t + 256] + be2[t + 256];
}

__global__ __launch_bounds__(256)
void ln_kernel(const float* __restrict__ a, const float* __restrict__ g,
               const float* __restrict__ be, float* __restrict__ out) {
    __shared__ float sm[8];
    int row = blockIdx.x, t = threadIdx.x;
    const float* ar = a + (size_t)row * DMODEL;
    float x0 = ar[t], x1 = ar[t + 256];
    float mean = block_reduce_sum(x0 + x1, sm) * (1.0f / DMODEL);
    float d0 = x0 - mean, d1 = x1 - mean;
    float var = block_reduce_sum(d0 * d0 + d1 * d1, sm) * (1.0f / DMODEL);
    float rs = rsqrtf(var + 1e-5f);
    float* o = out + (size_t)row * DMODEL;
    o[t]       = d0 * rs * g[t]       + be[t];
    o[t + 256] = d1 * rs * g[t + 256] + be[t + 256];
}

// =====================================================================
// cat2: scatter (256,H,25,25) attn into diagonal blocks of (B,H,100,100)
// =====================================================================
__global__ __launch_bounds__(256)
void cat2_kernel(const float* __restrict__ s2, float* __restrict__ out) {
    size_t idx = (size_t)blockIdx.x * blockDim.x + threadIdx.x;
    if (idx >= (size_t)ATTN1) return;
    int j = (int)(idx % WIN); size_t t = idx / WIN;
    int i = (int)(t % WIN);   t /= WIN;
    int h = (int)(t % NHEAD);
    int b = (int)(t / NHEAD);
    int bi = i / 25, bj = j / 25;
    float v = 0.0f;
    if (bi == bj)
        v = s2[((((size_t)(b * 4 + bi) * NHEAD + h) * 25 + (i % 25)) * 25) + (j % 25)];
    out[idx] = v;
}

// =====================================================================
// Host orchestration
// =====================================================================
struct EncParams {
    const float *wq, *bq, *wk, *bk, *wv, *bv, *wo, *bo;
    const float *w1, *b1, *w2, *b2;
    const float *n1g, *n1b, *n2g, *n2b, *eng, *enb;
};

extern "C" void kernel_launch(void* const* d_in, const int* in_sizes, int n_in,
                              void* d_out, int out_size, void* d_ws, size_t ws_size,
                              hipStream_t stream) {
    (void)in_sizes; (void)n_in; (void)out_size; (void)ws_size;
    int pi = 0;
    auto in = [&](void) -> const float* { return (const float*)d_in[pi++]; };

    const float* x_in  = in();
    const float* emb_w = in();
    EncParams enc1[3], enc2[3];
    auto read_layer = [&](EncParams& L) {
        L.wq = in(); L.bq = in(); L.wk = in(); L.bk = in();
        L.wv = in(); L.bv = in(); L.wo = in(); L.bo = in();
        L.w1 = in(); L.b1 = in(); L.w2 = in(); L.b2 = in();
        L.n1g = in(); L.n1b = in(); L.n2g = in(); L.n2b = in();
        L.eng = in(); L.enb = in();
    };
    for (int l = 0; l < 3; ++l) read_layer(enc1[l]);
    for (int l = 0; l < 3; ++l) read_layer(enc2[l]);
    const float *gws[3], *gbs[3];
    for (int l = 0; l < 3; ++l) { gws[l] = in(); gbs[l] = in(); }
    const float *lw[3], *lb[3], *lng[3], *lnb[3];
    for (int l = 0; l < 3; ++l) { lw[l] = in(); lb[l] = in(); lng[l] = in(); lnb[l] = in(); }
    const float* final_ng = in();
    const float* final_nb = in();
    const float* proj_w   = in();
    const float* proj_b   = in();

    // ---- workspace carve ----
    float* W = (float*)d_ws;
    size_t off = 0;
    auto wsbuf = [&](size_t n) { float* p = W + off; off += n; return p; };
    float* ENC = wsbuf((size_t)MTOK * DMODEL);
    float* Qb  = wsbuf((size_t)MTOK * DMODEL);
    float* Kb  = wsbuf((size_t)MTOK * DMODEL);
    float* Vb  = wsbuf((size_t)MTOK * DMODEL);
    float* T1  = wsbuf((size_t)MTOK * DMODEL);
    float* CC  = wsbuf((size_t)MTOK * 2 * DMODEL);
    float* SIG = wsbuf((size_t)NB_B * NHEAD * WIN);
    float* S2a = wsbuf((size_t)256 * NHEAD * 25 * 25);
    float* S2b = wsbuf((size_t)256 * NHEAD * 25 * 25);

    // ---- output carve (tuple flattened in return order) ----
    float* out = (float*)d_out;
    float* OUTP = out;                                   // (64,100,38)
    float *SER[3], *PRI[3], *SGM[3], *CAT[3];
    size_t base = (size_t)MTOK * C_OUT;
    for (int l = 0; l < 3; ++l) SER[l] = out + base + (size_t)l * ATTN1;
    for (int l = 0; l < 3; ++l) PRI[l] = out + base + (size_t)(3 + l) * ATTN1;
    for (int l = 0; l < 3; ++l) SGM[l] = out + base + (size_t)(6 + l) * ATTN1;
    for (int l = 0; l < 3; ++l) CAT[l] = out + base + (size_t)(9 + l) * ATTN1;

    const dim3 gemm_grid(MTOK / GBM, DMODEL / GBN);      // (25, 8)

    auto run_block = [&](const EncParams& L, int p, const float* pre,
                         float* attnOut, float* eOut, int eStride) {
        int nb = MTOK / p;
        gemm_bias_wmma<0><<<gemm_grid, 256, 0, stream>>>(ENC, L.wq, L.bq, Qb, MTOK, DMODEL, DMODEL);
        gemm_bias_wmma<0><<<gemm_grid, 256, 0, stream>>>(ENC, L.wk, L.bk, Kb, MTOK, DMODEL, DMODEL);
        gemm_bias_wmma<0><<<gemm_grid, 256, 0, stream>>>(ENC, L.wv, L.bv, Vb, MTOK, DMODEL, DMODEL);
        size_t stot = (size_t)nb * NHEAD * p * p;
        scores_kernel<<<cdiv(stot, 256), 256, 0, stream>>>(Qb, Kb, pre, attnOut, p, nb, 0.125f);
        int rows = nb * NHEAD * p;
        softmax_kernel<<<cdiv((size_t)rows, 8), 256, 0, stream>>>(attnOut, rows, p);
        attn_apply_kernel<<<cdiv((size_t)MTOK * DMODEL, 256), 256, 0, stream>>>(attnOut, Vb, T1, p);
        gemm_bias_wmma<0><<<gemm_grid, 256, 0, stream>>>(T1, L.wo, L.bo, Qb, MTOK, DMODEL, DMODEL);
        add_ln_kernel<<<MTOK, 256, 0, stream>>>(ENC, Qb, L.n1g, L.n1b, Kb, DMODEL);   // x2
        gemm_bias_wmma<2><<<gemm_grid, 256, 0, stream>>>(Kb, L.w1, L.b1, Vb, MTOK, DMODEL, DMODEL); // gelu
        gemm_bias_wmma<0><<<gemm_grid, 256, 0, stream>>>(Vb, L.w2, L.b2, Qb, MTOK, DMODEL, DMODEL);
        add_ln2_kernel<<<MTOK, 256, 0, stream>>>(Kb, Qb, L.n2g, L.n2b, L.eng, L.enb, eOut, eStride);
    };

    // ---- embedding ----
    embed_kernel<<<cdiv((size_t)MTOK * DMODEL, 256), 256, 0, stream>>>(x_in, emb_w, ENC);

    float* s2_prev = S2a;
    float* s2_cur  = S2b;
    for (int l = 0; l < 3; ++l) {
        // Gaussian prior + sigma straight into d_out
        gauss_sig_kernel<<<cdiv((size_t)NB_B * NHEAD * WIN, 256), 256, 0, stream>>>(ENC, gws[l], gbs[l], SIG);
        prior_sigma_kernel<<<cdiv((size_t)ATTN1, 256), 256, 0, stream>>>(SIG, PRI[l], SGM[l]);

        // branch 1: p=100, attn written directly into series output, prev series is `pre`
        run_block(enc1[l], 100, (l ? SER[l - 1] : nullptr), SER[l], CC, 2 * DMODEL);
        // branch 2: p=25, s2 ping-pong in workspace
        run_block(enc2[l], 25, (l ? s2_prev : nullptr), s2_cur, CC + DMODEL, 2 * DMODEL);
        cat2_kernel<<<cdiv((size_t)ATTN1, 256), 256, 0, stream>>>(s2_cur, CAT[l]);
        { float* t = s2_prev; s2_prev = s2_cur; s2_cur = t; }

        // lin: relu(cat @ W + b) then LN -> new enc
        gemm_bias_wmma<1><<<gemm_grid, 256, 0, stream>>>(CC, lw[l], lb[l], T1, MTOK, DMODEL, 2 * DMODEL);
        ln_kernel<<<MTOK, 256, 0, stream>>>(T1, lng[l], lnb[l], ENC);
    }

    // final LN + projection (N=38, guarded WMMA GEMM)
    ln_kernel<<<MTOK, 256, 0, stream>>>(ENC, final_ng, final_nb, T1);
    gemm_bias_wmma<0><<<dim3(MTOK / GBM, cdiv(C_OUT, GBN)), 256, 0, stream>>>(
        T1, proj_w, proj_b, OUTP, MTOK, C_OUT, DMODEL);
}
